// HyperLinear_88450556494407
// MI455X (gfx1250) — matrix operand order repaired and checked by reading, not compiled, not run on hardware
//
#include <hip/hip_runtime.h>

// ---------------------------------------------------------------------------
// HyperLinear (octonion block Linear) as one dense GEMM on CDNA5 WMMA.
//   y[n, o*512+g] = sum_j C[o,j] * sum_f W[A[o,j],g,f] * x[n, j*512+f] + bg[o,g]
// M=8192, N=4096, K=4096, fp32 in/out, bf16x3 split-precision WMMA.
// Fast path: pre-split x/W to bf16 hi/lo in d_ws once, then async-to-LDS
// double-buffered GEMM (pure DMA staging, no VALU conversion in the loop).
// ---------------------------------------------------------------------------

typedef __attribute__((ext_vector_type(16))) __bf16 v16bf;
typedef __attribute__((ext_vector_type(8)))  float  v8f;
typedef __attribute__((ext_vector_type(4)))  unsigned int u32x4;

union Frag { v16bf v; u32x4 q[2]; };

// ---- compile-time Cayley-Dickson routing tables (A = component idx, C = sign)
struct Tables { int a[8][8]; int c[8][8]; };

constexpr Tables make_tables() {
    int s[8][8] = {}, k[8][8] = {};
    s[0][0] = 1; k[0][0] = 0;
    for (int m = 1; m < 8; m *= 2) {
        int s2[8][8] = {}, k2[8][8] = {};
        for (int a = 0; a < m; ++a)
            for (int c = 0; c < m; ++c) {
                int cs = (c == 0) ? 1 : -1;
                s2[a][c]         = s[a][c];          k2[a][c]         = k[a][c];
                s2[a][c + m]     = s[c][a];          k2[a][c + m]     = k[c][a] + m;
                s2[a + m][c]     = s[a][c] * cs;     k2[a + m][c]     = k[a][c] + m;
                s2[a + m][c + m] = -(s[c][a] * cs);  k2[a + m][c + m] = k[c][a];
            }
        for (int a = 0; a < 2 * m; ++a)
            for (int c = 0; c < 2 * m; ++c) { s[a][c] = s2[a][c]; k[a][c] = k2[a][c]; }
    }
    Tables t = {};
    for (int j = 0; j < 8; ++j)
        for (int a = 0; a < 8; ++a) { int i = k[a][j]; t.a[i][j] = a; t.c[i][j] = s[a][j]; }
    return t;
}

__constant__ Tables dTAB = make_tables();

// ---- fp32 -> (hi, lo) bf16 split: v ~= hi + lo, rel err ~2^-17
__device__ __forceinline__ void bsplit(float v, unsigned short& h, unsigned short& l) {
    unsigned int u = __float_as_uint(v);
    h = (unsigned short)(u >> 16);                       // truncate to bf16
    float hf = __uint_as_float((unsigned int)h << 16);
    unsigned int r = __float_as_uint(v - hf);            // exact residual
    l = (unsigned short)((r + 0x8000u) >> 16);           // RNE-ish low part
}

// ---- CDNA5 async copy: 16B global -> LDS, tracked by ASYNCcnt (no VGPR data)
__device__ __forceinline__ void async_b128(unsigned lds_addr, const void* gaddr) {
    asm volatile("global_load_async_to_lds_b128 %0, %1, off"
                 :: "v"(lds_addr), "v"((unsigned long long)(uintptr_t)gaddr)
                 : "memory");
}

#define TM    128
#define TN    128
#define TK    32
#define PITCH 40          // ushorts per LDS row (80B: 16B-aligned, low-conflict)
#define NSTEP (4096 / TK)

// Per-buffer LDS layout (ushort units): XH 0, XL 5120, WH 10240, WL 15360
#define BUFU  20480       // ushorts per buffer
#define BUFB  40960       // bytes per buffer
#define LDSB  (2 * BUFB)  // 80 KB double-buffered

// ===========================================================================
// FAST PATH: pre-split bf16 operands, async-to-LDS double-buffered GEMM
// ===========================================================================
__global__ __launch_bounds__(256)
void hl_gemm_async(const unsigned short* __restrict__ xh, const unsigned short* __restrict__ xl,
                   const unsigned short* __restrict__ wh, const unsigned short* __restrict__ wl,
                   const float* __restrict__ bg, float* __restrict__ y) {
    extern __shared__ unsigned short smem[];

    const int tid  = threadIdx.x;
    const int lane = tid & 31;
    const int wave = tid >> 5;
    const int wm   = wave >> 2;     // 0..1 -> 64 rows
    const int wn   = wave & 3;      // 0..3 -> 32 cols

    const int og0 = blockIdx.x * TN;
    const int m0  = blockIdx.y * TM;
    const int o   = og0 >> 9;
    const int g0  = og0 & 511;

    const unsigned ldsbase = (unsigned)(uintptr_t)smem;

    // stage `step` -> buffer `buf`: 8 async b128 per thread (512 chunks/tile)
    auto issue = [&](int step, int buf) {
        const int k0 = step * TK;
        const int j  = k0 >> 9;
        const int f0 = k0 & 511;
        const size_t wbase = ((size_t)dTAB.a[o][j] << 18);   // component * 512*512
#pragma unroll
        for (int c = 0; c < 2; ++c) {
            const int idx  = tid + c * 256;     // 0..511
            const int trow = idx >> 2;          // 0..127
            const int cb   = (idx & 3) << 4;    // byte offset in 64B row
            const unsigned lrow = ldsbase + (unsigned)(buf * BUFB + trow * 80 + cb);
            const size_t xoff = ((size_t)(m0 + trow) * 4096 + k0) * 2 + cb;
            const size_t woff = (wbase + (size_t)(g0 + trow) * 512 + f0) * 2 + cb;
            async_b128(lrow +     0u, (const char*)xh + xoff);
            async_b128(lrow + 10240u, (const char*)xl + xoff);
            async_b128(lrow + 20480u, (const char*)wh + woff);
            async_b128(lrow + 30720u, (const char*)wl + woff);
        }
    };

    v8f acc[4][2];
#pragma unroll
    for (int mi = 0; mi < 4; ++mi)
#pragma unroll
        for (int ni = 0; ni < 2; ++ni)
#pragma unroll
            for (int r = 0; r < 8; ++r) acc[mi][ni][r] = 0.0f;

    issue(0, 0);

    for (int s = 0; s < NSTEP; ++s) {
        const int buf = s & 1;
        if (s + 1 < NSTEP) {
            issue(s + 1, buf ^ 1);
            asm volatile("s_wait_asynccnt 8" ::: "memory");   // stage s done
        } else {
            asm volatile("s_wait_asynccnt 0" ::: "memory");
        }
        __syncthreads();

        const unsigned short* bXh = smem + buf * BUFU;
        const unsigned short* bXl = bXh + 5120;
        const unsigned short* bWh = bXh + 10240;
        const unsigned short* bWl = bXh + 15360;

        Frag ah[4], al[4], bh[2], bl[2];
        const int mrow = lane & 15;
        const int ak   = (lane >> 4) << 3;    // A: upper lanes hold K+8 / K+24
#pragma unroll
        for (int mi = 0; mi < 4; ++mi) {
            const int base = (wm * 64 + mi * 16 + mrow) * PITCH + ak;
            ah[mi].q[0] = *(const u32x4*)&bXh[base];
            ah[mi].q[1] = *(const u32x4*)&bXh[base + 16];
            al[mi].q[0] = *(const u32x4*)&bXl[base];
            al[mi].q[1] = *(const u32x4*)&bXl[base + 16];
        }
        const int bk = (lane >> 4) << 3;      // B: upper lanes hold K+8 / K+24
#pragma unroll
        for (int ni = 0; ni < 2; ++ni) {
            const int base = (wn * 32 + ni * 16 + mrow) * PITCH + bk;
            bh[ni].q[0] = *(const u32x4*)&bWh[base];
            bh[ni].q[1] = *(const u32x4*)&bWh[base + 16];
            bl[ni].q[0] = *(const u32x4*)&bWl[base];
            bl[ni].q[1] = *(const u32x4*)&bWl[base + 16];
        }

        // sign(o, j): uniform; negate B fragments (bf16 WMMA NEG is C-only)
        if (dTAB.c[o][s >> 4] < 0) {
#pragma unroll
            for (int ni = 0; ni < 2; ++ni) {
                bh[ni].q[0] ^= 0x80008000u; bh[ni].q[1] ^= 0x80008000u;
                bl[ni].q[0] ^= 0x80008000u; bl[ni].q[1] ^= 0x80008000u;
            }
        }

        // bf16x3: hi*hi + hi*lo + lo*hi, fp32 accumulate
#pragma unroll
        for (int mi = 0; mi < 4; ++mi)
#pragma unroll
            for (int ni = 0; ni < 2; ++ni) {
                v8f c = acc[mi][ni];
                c = __builtin_amdgcn_wmma_f32_16x16x32_bf16(
                        false, ah[mi].v, false, bh[ni].v, (short)0, c, false, false);
                c = __builtin_amdgcn_wmma_f32_16x16x32_bf16(
                        false, ah[mi].v, false, bl[ni].v, (short)0, c, false, false);
                c = __builtin_amdgcn_wmma_f32_16x16x32_bf16(
                        false, al[mi].v, false, bh[ni].v, (short)0, c, false, false);
                acc[mi][ni] = c;
            }
        __syncthreads();
    }

    // epilogue: C/D layout: VGPR r -> (M = r + 8*(lane>=16), N = lane&15)
    const int coln   = lane & 15;
    const int rowsel = (lane >> 4) << 3;
#pragma unroll
    for (int ni = 0; ni < 2; ++ni) {
        const int col  = og0 + wn * 32 + ni * 16 + coln;
        const float bv = bg[col];
#pragma unroll
        for (int mi = 0; mi < 4; ++mi) {
            const int rowb = m0 + wm * 64 + mi * 16 + rowsel;
#pragma unroll
            for (int r = 0; r < 8; ++r)
                y[(size_t)(rowb + r) * 4096 + col] = acc[mi][ni][r] + bv;
        }
    }
}

// ---- pre-split: 8 fp32 -> 8 (hi,lo) bf16 per thread, packed 16B stores
__global__ __launch_bounds__(256)
void hl_split(const float* __restrict__ src, unsigned short* __restrict__ h16,
              unsigned short* __restrict__ l16) {
    const size_t i  = (size_t)blockIdx.x * 256 + threadIdx.x;   // 8 elems each
    const float4 a = *(const float4*)(src + i * 8);
    const float4 b = *(const float4*)(src + i * 8 + 4);
    unsigned short h[8], l[8];
    bsplit(a.x, h[0], l[0]); bsplit(a.y, h[1], l[1]);
    bsplit(a.z, h[2], l[2]); bsplit(a.w, h[3], l[3]);
    bsplit(b.x, h[4], l[4]); bsplit(b.y, h[5], l[5]);
    bsplit(b.z, h[6], l[6]); bsplit(b.w, h[7], l[7]);
    uint4 hv, lv;
    hv.x = (unsigned)h[0] | ((unsigned)h[1] << 16); hv.y = (unsigned)h[2] | ((unsigned)h[3] << 16);
    hv.z = (unsigned)h[4] | ((unsigned)h[5] << 16); hv.w = (unsigned)h[6] | ((unsigned)h[7] << 16);
    lv.x = (unsigned)l[0] | ((unsigned)l[1] << 16); lv.y = (unsigned)l[2] | ((unsigned)l[3] << 16);
    lv.z = (unsigned)l[4] | ((unsigned)l[5] << 16); lv.w = (unsigned)l[6] | ((unsigned)l[7] << 16);
    *(uint4*)(h16 + i * 8) = hv;
    *(uint4*)(l16 + i * 8) = lv;
}

// ===========================================================================
// FALLBACK (small d_ws): round-1 kernel, fp32->bf16 split done in-kernel
// ===========================================================================
__global__ __launch_bounds__(256)
void hl_gemm(const float* __restrict__ x, const float* __restrict__ W,
             const float* __restrict__ bg, float* __restrict__ y) {
    __shared__ __align__(16) unsigned short sXh[TM * PITCH];
    __shared__ __align__(16) unsigned short sXl[TM * PITCH];
    __shared__ __align__(16) unsigned short sWh[TN * PITCH];
    __shared__ __align__(16) unsigned short sWl[TN * PITCH];

    const int tid  = threadIdx.x;
    const int lane = tid & 31;
    const int wave = tid >> 5;
    const int wm   = wave >> 2;
    const int wn   = wave & 3;

    const int og0 = blockIdx.x * TN;
    const int m0  = blockIdx.y * TM;
    const int o   = og0 >> 9;
    const int g0  = og0 & 511;

    int   aj[8];
    float sj[8];
#pragma unroll
    for (int j = 0; j < 8; ++j) { aj[j] = dTAB.a[o][j]; sj[j] = (float)dTAB.c[o][j]; }

    v8f acc[4][2];
#pragma unroll
    for (int mi = 0; mi < 4; ++mi)
#pragma unroll
        for (int ni = 0; ni < 2; ++ni)
#pragma unroll
            for (int r = 0; r < 8; ++r) acc[mi][ni][r] = 0.0f;

    for (int k0 = 0; k0 < 4096; k0 += TK) {
        const int   j    = k0 >> 9;
        const int   f0   = k0 & 511;
        const float sign = sj[j];
        const float* Wb  = W + ((size_t)aj[j] << 18);
#pragma unroll
        for (int i = 0; i < 4; ++i) {
            const int idx = tid + i * 256;
            const int row = idx >> 3;
            const int c4  = (idx & 7) << 2;
            const float4 xv = *(const float4*)(x + (size_t)(m0 + row) * 4096 + k0 + c4);
            const float4 wv = *(const float4*)(Wb + (size_t)(g0 + row) * 512 + f0 + c4);
            const int so = row * PITCH + c4;
            unsigned short h, l;
            bsplit(xv.x, h, l);        sXh[so + 0] = h; sXl[so + 0] = l;
            bsplit(xv.y, h, l);        sXh[so + 1] = h; sXl[so + 1] = l;
            bsplit(xv.z, h, l);        sXh[so + 2] = h; sXl[so + 2] = l;
            bsplit(xv.w, h, l);        sXh[so + 3] = h; sXl[so + 3] = l;
            bsplit(sign * wv.x, h, l); sWh[so + 0] = h; sWl[so + 0] = l;
            bsplit(sign * wv.y, h, l); sWh[so + 1] = h; sWl[so + 1] = l;
            bsplit(sign * wv.z, h, l); sWh[so + 2] = h; sWl[so + 2] = l;
            bsplit(sign * wv.w, h, l); sWh[so + 3] = h; sWl[so + 3] = l;
        }
        __syncthreads();

        Frag ah[4], al[4], bh[2], bl[2];
        const int mrow = lane & 15;
        const int ak   = (lane >> 4) << 3;
#pragma unroll
        for (int mi = 0; mi < 4; ++mi) {
            const int base = (wm * 64 + mi * 16 + mrow) * PITCH + ak;
            ah[mi].q[0] = *(const u32x4*)&sXh[base];
            ah[mi].q[1] = *(const u32x4*)&sXh[base + 16];
            al[mi].q[0] = *(const u32x4*)&sXl[base];
            al[mi].q[1] = *(const u32x4*)&sXl[base + 16];
        }
        const int bk = (lane >> 4) << 3;
#pragma unroll
        for (int ni = 0; ni < 2; ++ni) {
            const int base = (wn * 32 + ni * 16 + mrow) * PITCH + bk;
            bh[ni].q[0] = *(const u32x4*)&sWh[base];
            bh[ni].q[1] = *(const u32x4*)&sWh[base + 16];
            bl[ni].q[0] = *(const u32x4*)&sWl[base];
            bl[ni].q[1] = *(const u32x4*)&sWl[base + 16];
        }
#pragma unroll
        for (int mi = 0; mi < 4; ++mi)
#pragma unroll
            for (int ni = 0; ni < 2; ++ni) {
                v8f c = acc[mi][ni];
                c = __builtin_amdgcn_wmma_f32_16x16x32_bf16(
                        false, ah[mi].v, false, bh[ni].v, (short)0, c, false, false);
                c = __builtin_amdgcn_wmma_f32_16x16x32_bf16(
                        false, ah[mi].v, false, bl[ni].v, (short)0, c, false, false);
                c = __builtin_amdgcn_wmma_f32_16x16x32_bf16(
                        false, al[mi].v, false, bh[ni].v, (short)0, c, false, false);
                acc[mi][ni] = c;
            }
        __syncthreads();
    }

    const int coln   = lane & 15;
    const int rowsel = (lane >> 4) << 3;
#pragma unroll
    for (int ni = 0; ni < 2; ++ni) {
        const int col  = og0 + wn * 32 + ni * 16 + coln;
        const float bv = bg[col];
#pragma unroll
        for (int mi = 0; mi < 4; ++mi) {
            const int rowb = m0 + wm * 64 + mi * 16 + rowsel;
#pragma unroll
            for (int r = 0; r < 8; ++r)
                y[(size_t)(rowb + r) * 4096 + col] = acc[mi][ni][r] + bv;
        }
    }
}

// ---- bias prep: bg[o*512+g] = sum_j C[o,j] * b[A[o,j], g]
__global__ __launch_bounds__(256)
void hl_bias(const float* __restrict__ b, float* __restrict__ bg) {
    const int og = blockIdx.x * 256 + threadIdx.x;
    const int o  = og >> 9;
    const int g  = og & 511;
    float acc = 0.0f;
#pragma unroll
    for (int j = 0; j < 8; ++j)
        acc += (float)dTAB.c[o][j] * b[dTAB.a[o][j] * 512 + g];
    bg[og] = acc;
}

extern "C" void kernel_launch(void* const* d_in, const int* in_sizes, int n_in,
                              void* d_out, int out_size, void* d_ws, size_t ws_size,
                              hipStream_t stream) {
    (void)in_sizes; (void)n_in; (void)out_size;
    const float* x = (const float*)d_in[0];   // [8192, 4096]
    const float* W = (const float*)d_in[1];   // [8, 512, 512]
    const float* b = (const float*)d_in[2];   // [8, 512]
    float* y  = (float*)d_out;                // [8192, 4096]
    float* bg = (float*)d_ws;                 // 4096 floats

    hl_bias<<<16, 256, 0, stream>>>(b, bg);

    const size_t NX = (size_t)8192 * 4096;    // x elements
    const size_t NW = (size_t)8 * 512 * 512;  // W elements
    const size_t need = 16384 + 2 * NX * 2 + 2 * NW * 2;

    dim3 grid(4096 / TN, 8192 / TM);          // (32, 64)
    if (ws_size >= need) {
        unsigned short* xh = (unsigned short*)((char*)d_ws + 16384);
        unsigned short* xl = xh + NX;
        unsigned short* wh = xl + NX;
        unsigned short* wl = wh + NW;
        hl_split<<<(int)(NX / (256 * 8)), 256, 0, stream>>>(x, xh, xl);   // 16384 blocks
        hl_split<<<(int)(NW / (256 * 8)), 256, 0, stream>>>(W, wh, wl);   // 1024 blocks
        hipFuncSetAttribute((const void*)hl_gemm_async,
                            hipFuncAttributeMaxDynamicSharedMemorySize, LDSB);
        hl_gemm_async<<<grid, 256, LDSB, stream>>>(xh, xl, wh, wl, bg, y);
    } else {
        hl_gemm<<<grid, 256, 0, stream>>>(x, W, bg, y);
    }
}
